// cross_GAE_VAE_87342454931668
// MI455X (gfx1250) — matrix operand order repaired and checked by reading, not compiled, not run on hardware
//
#include <hip/hip_runtime.h>
#include <hip/hip_bf16.h>
#include <stdint.h>
#include <stddef.h>

typedef _Float16 h16;
typedef __attribute__((ext_vector_type(16))) _Float16 v16h;
typedef __attribute__((ext_vector_type(8)))  float    v8f;

namespace {
constexpr int Nn  = 20000;
constexpr int Ee  = 320000;
constexpr int DSd = 2000;
constexpr int DRd = 1000;
constexpr int DTd = 1000;
constexpr int HID = 512;
constexpr int LATd= 64;
constexpr int NCd = 20;
constexpr float NOISEF = 0.1f;

__host__ __device__ constexpr int cdiv(int a, int b){ return (a + b - 1) / b; }
__host__ __device__ constexpr int rup (int a, int b){ return cdiv(a, b) * b; }
} // namespace

// ----------------------------- device helpers ------------------------------

__device__ __forceinline__ uint32_t pcg_hash(uint32_t v){
  v = v * 747796405u + 2891336453u;
  v = ((v >> ((v >> 28u) + 4u)) ^ v) * 277803737u;
  return (v >> 22u) ^ v;
}
__device__ __forceinline__ float u01(uint32_t h){
  return ((h >> 8) + 1u) * (1.0f / 16777217.0f);
}
__device__ __forceinline__ float gaussrnd(uint32_t seed, uint32_t idx){
  uint32_t h1 = pcg_hash(seed ^ (idx * 2654435761u));
  uint32_t h2 = pcg_hash(h1 + 0x9E3779B9u);
  float r = sqrtf(-2.0f * logf(u01(h1)));
  return r * cosf(6.28318530717958647f * u01(h2));
}
__device__ __forceinline__ int   f2ord(float f){ int i = __float_as_int(f); return i >= 0 ? i : (i ^ 0x7FFFFFFF); }
__device__ __forceinline__ float ord2f(int i){ return __int_as_float(i >= 0 ? i : (i ^ 0x7FFFFFFF)); }

// CDNA5 async copy: 16B from per-lane global addr into per-lane LDS addr.
// Tracked by ASYNCcnt (s_wait_asynccnt).
__device__ __forceinline__ void async_ld_b128(void* lds_ptr, const void* gptr){
  uint32_t lo = (uint32_t)(uintptr_t)lds_ptr;   // low 32 bits of flat LDS addr = LDS offset
  asm volatile("global_load_async_to_lds_b128 %0, %1, off"
               :: "v"(lo), "v"((uint64_t)(uintptr_t)gptr)
               : "memory");
}
__device__ __forceinline__ void wait_async_le3(){
  asm volatile("s_wait_asynccnt 0x3" ::: "memory");
}
__device__ __forceinline__ void wait_async_le0(){
  asm volatile("s_wait_asynccnt 0x0" ::: "memory");
}

// ----------------------------- small kernels -------------------------------

__global__ void fill_f32(float* p, float v, int n){
  int i = blockIdx.x * blockDim.x + threadIdx.x;
  if (i < n) p[i] = v;
}
__global__ void fill_i32(int* p, int v, int n){
  int i = blockIdx.x * blockDim.x + threadIdx.x;
  if (i < n) p[i] = v;
}

// x (n x cols_in) [+ optional flag col] + noise -> f16, zero-padded to ldo
__global__ void noise_cast(const float* __restrict__ x, h16* __restrict__ out,
                           int n, int cols_in, int flag_cols, float flagval,
                           int ldo, uint32_t seed){
  size_t i = (size_t)blockIdx.x * blockDim.x + threadIdx.x;
  size_t total = (size_t)n * ldo;
  if (i >= total) return;
  int row = (int)(i / ldo), c = (int)(i % ldo);
  int cin = cols_in + flag_cols;
  float v = 0.0f;
  if (c < cin){
    float base = (c < cols_in) ? x[(size_t)row * cols_in + c] : flagval;
    v = base + NOISEF * gaussrnd(seed, (uint32_t)(row * cin + c));
  }
  out[i] = (h16)v;
}

// W (srcR x srcC) -> out (Rb x Cb) f16, optional transpose, zero-pad
__global__ void cast_pad(const float* __restrict__ W, h16* __restrict__ out,
                         int Rb, int Cb, int srcR, int srcC, int transpose){
  size_t i = (size_t)blockIdx.x * blockDim.x + threadIdx.x;
  size_t total = (size_t)Rb * Cb;
  if (i >= total) return;
  int r = (int)(i / Cb), c = (int)(i % Cb);
  float v = 0.0f;
  if (!transpose){ if (r < srcR && c < srcC) v = W[(size_t)r * srcC + c]; }
  else           { if (c < srcR && r < srcC) v = W[(size_t)c * srcC + r]; }
  out[i] = (h16)v;
}

// f32 -> f16, optional ELU
__global__ void cast_act(const float* __restrict__ in, h16* __restrict__ out,
                         size_t n, int apply_elu){
  size_t i = (size_t)blockIdx.x * blockDim.x + threadIdx.x;
  if (i >= n) return;
  float v = in[i];
  if (apply_elu) v = v > 0.0f ? v : (expf(v) - 1.0f);
  out[i] = (h16)v;
}

__global__ void expk(const float* __restrict__ in, float* __restrict__ out, int n){
  int i = blockIdx.x * blockDim.x + threadIdx.x;
  if (i < n) out[i] = expf(in[i]);
}

__global__ void concat_cast(const float* __restrict__ a, const float* __restrict__ b,
                            h16* __restrict__ out, int n, int lat){
  size_t i = (size_t)blockIdx.x * blockDim.x + threadIdx.x;
  size_t total = (size_t)n * 2 * lat;
  if (i >= total) return;
  int row = (int)(i / (2 * lat)), c = (int)(i % (2 * lat));
  float v = (c < lat) ? a[(size_t)row * lat + c] : b[(size_t)row * lat + (c - lat)];
  out[i] = (h16)v;
}

__global__ void reparam(const float* __restrict__ mean, const float* __restrict__ lv,
                        float* __restrict__ z, h16* __restrict__ zh,
                        size_t n, uint32_t seed){
  size_t i = (size_t)blockIdx.x * blockDim.x + threadIdx.x;
  if (i >= n) return;
  float stdv = sqrtf(expf(0.5f * lv[i]) + 1e-8f);
  float zz = mean[i] + gaussrnd(seed, (uint32_t)i) * stdv;
  z[i]  = zz;
  zh[i] = (h16)zz;
}

// ----------------------------- attention kernels ---------------------------

__global__ __launch_bounds__(256)
void dot_attn(const float* __restrict__ H, const float* __restrict__ aw_s,
              const float* __restrict__ aw_d, float* __restrict__ a_s,
              float* __restrict__ a_d, int n, int hid){
  int wave = threadIdx.x >> 5, lane = threadIdx.x & 31;
  int node = blockIdx.x * 8 + wave;
  if (node >= n) return;
  float ss = 0.0f, sd = 0.0f;
  for (int c = lane; c < hid; c += 32){
    float h = H[(size_t)node * hid + c];
    ss += h * aw_s[c];
    sd += h * aw_d[c];
  }
  for (int o = 16; o; o >>= 1){
    ss += __shfl_xor(ss, o, 32);
    sd += __shfl_xor(sd, o, 32);
  }
  if (lane == 0){ a_s[node] = ss; a_d[node] = sd; }
}

__global__ void edge_max(const float* __restrict__ a_s, const float* __restrict__ a_d,
                         const int* __restrict__ src, const int* __restrict__ dst,
                         float* __restrict__ logit, int* __restrict__ m_int, int E){
  int e = blockIdx.x * blockDim.x + threadIdx.x;
  if (e >= E) return;
  float l = a_s[src[e]] + a_d[dst[e]];
  l = l > 0.0f ? l : 0.2f * l;           // leaky_relu(0.2)
  logit[e] = l;
  atomicMax(&m_int[dst[e]], f2ord(l));
}

__global__ void edge_exp(const float* __restrict__ logit, const int* __restrict__ dst,
                         const int* __restrict__ m_int, float* __restrict__ expv,
                         float* __restrict__ ssum, int E){
  int e = blockIdx.x * blockDim.x + threadIdx.x;
  if (e >= E) return;
  int d = dst[e];
  float ev = expf(logit[e] - ord2f(m_int[d]));
  expv[e] = ev;
  atomicAdd(&ssum[d], ev);
}

__global__ __launch_bounds__(256)
void edge_aggr(const float* __restrict__ H, const int* __restrict__ src,
               const int* __restrict__ dst, const float* __restrict__ expv,
               const float* __restrict__ ssum, float* __restrict__ outb,
               int E, int hid){
  int e = blockIdx.x;
  if (e >= E) return;
  int s = src[e], d = dst[e];
  float alpha = expv[e] / (ssum[d] + 1e-16f);
  for (int c = threadIdx.x; c < hid; c += blockDim.x)
    atomicAdd(&outb[(size_t)d * hid + c], H[(size_t)s * hid + c] * alpha);
}

// ----------------------------- WMMA GEMM -----------------------------------
// C[M,Nc] = A[M,Kp](f16, row-major) x Bt[Nc,Kp](f16, K-contiguous) + epilogue.
// 256 threads = 8 waves; block tile 128x64; wave tile 32x32 = 2x2 WMMA
// (v_wmma_f32_16x16x32_f16). K step 32. Double-buffered LDS staged via
// global_load_async_to_lds_b128 (ASYNCcnt pipeline). Out-of-range rows/cols
// are clamped at load (they only feed never-stored C elements).

union FragAB { v16h h; unsigned u[8]; };
union FragC  { v8f  f; float    x[8]; };

__global__ __launch_bounds__(256)
void wmma_gemm(const h16* __restrict__ A, int lda,
               const h16* __restrict__ Bt, int ldb,
               float* __restrict__ C, int ldc,
               int M, int Nc, int Kp, int ep,
               const float* __restrict__ bias, const float* __restrict__ bn_g,
               const float* __restrict__ bn_b, const float* __restrict__ bn_m,
               const float* __restrict__ bn_v){
  __shared__ h16 As[2][128 * 32];   // [row][k]
  __shared__ h16 Bs[2][64 * 32];    // [n][k]

  const int tid  = threadIdx.x;
  const int lane = tid & 31;
  const int wave = tid >> 5;
  const int wr = wave & 3;          // 32-row group
  const int wc = wave >> 2;         // 32-col group
  const int blockM = blockIdx.x * 128;
  const int blockN = blockIdx.y * 64;

  // per-thread staging coordinates (16B chunks, rows are 64B)
  const int arow0 = tid >> 2;                 // A chunk 0: rows 0..63
  const int arow1 = (tid + 256) >> 2;         // A chunk 1: rows 64..127
  const int acol  = (tid & 3) * 8;            // half offset within row
  int agr0 = blockM + arow0; if (agr0 >= M) agr0 = M - 1;
  int agr1 = blockM + arow1; if (agr1 >= M) agr1 = M - 1;
  const int brow  = tid >> 2;                 // B rows 0..63
  int bgn = blockN + brow; if (bgn >= Nc) bgn = Nc - 1;

  auto issue = [&](int buf, int k0){
    async_ld_b128(&As[buf][arow0 * 32 + acol], A  + (size_t)agr0 * lda + k0 + acol);
    async_ld_b128(&As[buf][arow1 * 32 + acol], A  + (size_t)agr1 * lda + k0 + acol);
    async_ld_b128(&Bs[buf][brow  * 32 + acol], Bt + (size_t)bgn  * ldb + k0 + acol);
  };

  FragC acc[2][2];
  for (int i = 0; i < 2; ++i)
    for (int j = 0; j < 2; ++j)
      for (int r = 0; r < 8; ++r) acc[i][j].x[r] = 0.0f;

  const int mloc = lane & 15;
  const int half = lane >> 4;
  const int nk = Kp >> 5;

  issue(0, 0);
  for (int it = 0; it < nk; ++it){
    const int cur = it & 1;
    const bool hasNext = (it + 1) < nk;
    if (hasNext){ issue(1 - cur, (it + 1) * 32); wait_async_le3(); }
    else        { wait_async_le0(); }
    __syncthreads();                       // current tile visible to all waves

    FragAB fa[2], fb[2];
    for (int t = 0; t < 2; ++t){
      int mrow = wr * 32 + t * 16 + mloc;
      for (int j = 0; j < 8; ++j){
        int kk = (j < 4 ? 2 * j : 16 + 2 * (j - 4)) + 8 * half;
        fa[t].u[j] = *(const unsigned*)(&As[cur][mrow * 32 + kk]);
      }
    }
    for (int t = 0; t < 2; ++t){
      int ncol = wc * 32 + t * 16 + mloc;
      for (int j = 0; j < 8; ++j){
        int kk = (j < 4 ? 2 * j : 16 + 2 * (j - 4)) + 8 * half;
        fb[t].u[j] = *(const unsigned*)(&Bs[cur][ncol * 32 + kk]);
      }
    }
    for (int i = 0; i < 2; ++i)
      for (int j = 0; j < 2; ++j)
        acc[i][j].f = __builtin_amdgcn_wmma_f32_16x16x32_f16(
            false, fa[i].h, false, fb[j].h, (short)0, acc[i][j].f, false, false);
    __syncthreads();                       // done reading before next overwrite
  }

  // ---- store with fused epilogue ----
  for (int i = 0; i < 2; ++i){
    for (int j = 0; j < 2; ++j){
      int col = blockN + wc * 32 + j * 16 + mloc;
      if (col >= Nc) continue;
      for (int r = 0; r < 8; ++r){
        int row = blockM + wr * 32 + i * 16 + r + 8 * half;
        if (row >= M) continue;
        float v = acc[i][j].x[r];
        if (ep == 1 || ep == 2 || ep == 4) v += bias[col];
        if (ep == 2) v = fminf(10.0f, fmaxf(-10.0f, v));
        if (ep == 3) v = fmaxf(v, 0.0f) + log1pf(expf(-fabsf(v)));   // softplus
        if (ep == 4){
          v = bn_g[col] * (v - bn_m[col]) * rsqrtf(bn_v[col] + 1e-5f) + bn_b[col];
          const float sa = 1.6732632423543772f, ss = 1.0507009873554805f;
          v = v > 0.0f ? ss * v : ss * sa * (expf(v) - 1.0f);        // SELU
        }
        C[(size_t)row * ldc + col] = v;
      }
    }
  }
}

// ----------------------------- host orchestration --------------------------

namespace {

struct Scratch {
  h16 *Xh;
  h16 *BtXW1;   // W1^T : [HID][Kp1]      (x @ W1)
  h16 *BtMu;    // W1   : [dflag][HID]    (h3 @ W1^T)
  h16 *BtHid;   // W2^T : [LAT][HID]      (h1 @ W2)
  h16 *BtDec;   // W2   : [HID][LAT]      (z @ W2^T)
  h16 *BtMean, *BtLv;                     // mW^T, lvW^T : [LAT][LAT]
  float *H, *aggr;
  h16 *h1h;
  float *hidden;
  h16 *hiddenh, *zh;
  float *a_s, *a_d, *ssum;
  int   *m_int;
  float *logit, *expv;
  h16 *clsin, *clshh;
  float *clsh;
  h16 *cW1t, *cW2t;                       // cW1^T [128][128], cW2^T [NC][128]
};

struct GBlk {
  const float* x; int din; int addflag; float flagval;
  const float *W1, *a1s, *a1d, *W2, *mW, *mb, *lvW, *lvb, *lth;
  const int *src, *dst;
  float *o_mean, *o_lv, *o_mu, *o_theta, *o_z;
  uint32_t seed;
};

void launch_gemm(hipStream_t st, const h16* A, int lda, const h16* Bt, int ldb,
                 float* C, int ldc, int M, int Nc, int Kp, int ep,
                 const float* bias = nullptr, const float* g = nullptr,
                 const float* b = nullptr, const float* rm = nullptr,
                 const float* rv = nullptr){
  dim3 grid(cdiv(M, 128), cdiv(Nc, 64));
  wmma_gemm<<<grid, 256, 0, st>>>(A, lda, Bt, ldb, C, ldc, M, Nc, Kp, ep,
                                  bias, g, b, rm, rv);
}

void run_block(hipStream_t st, const Scratch& S, const GBlk& g){
  const int dflag = g.din + g.addflag;
  const int Kp1   = rup(dflag, 32);

  // noisy f16 input, padded to Kp1
  {
    size_t tot = (size_t)Nn * Kp1;
    noise_cast<<<cdiv((int)tot, 256), 256, 0, st>>>(
        g.x, S.Xh, Nn, g.din, g.addflag, g.flagval, Kp1, g.seed);
  }
  // stage f16 weights in Bt ([N][K], K-contiguous) layout
  auto pad = [&](const float* W, h16* out, int Rb, int Cb, int sR, int sC, int tr){
    size_t tot = (size_t)Rb * Cb;
    cast_pad<<<cdiv((int)tot, 256), 256, 0, st>>>(W, out, Rb, Cb, sR, sC, tr);
  };
  pad(g.W1,  S.BtXW1,  HID,  Kp1,  dflag, HID,  1);  // W1^T, K-padded
  pad(g.W1,  S.BtMu,   dflag,HID,  dflag, HID,  0);  // W1
  pad(g.W2,  S.BtHid,  LATd, HID,  HID,   LATd, 1);  // W2^T
  pad(g.W2,  S.BtDec,  HID,  LATd, HID,   LATd, 0);  // W2
  pad(g.mW,  S.BtMean, LATd, LATd, LATd,  LATd, 1);  // mW^T
  pad(g.lvW, S.BtLv,   LATd, LATd, LATd,  LATd, 1);  // lvW^T

  // H = x @ W1
  launch_gemm(st, S.Xh, Kp1, S.BtXW1, Kp1, S.H, HID, Nn, HID, Kp1, 0);
  // attention scalars
  dot_attn<<<cdiv(Nn, 8), 256, 0, st>>>(S.H, g.a1s, g.a1d, S.a_s, S.a_d, Nn, HID);
  // segment softmax
  fill_i32<<<cdiv(Nn, 256), 256, 0, st>>>(S.m_int, (int)0x807FFFFF, Nn); // ord(-inf)
  fill_f32<<<cdiv(Nn, 256), 256, 0, st>>>(S.ssum, 0.0f, Nn);
  edge_max<<<cdiv(Ee, 256), 256, 0, st>>>(S.a_s, S.a_d, g.src, g.dst, S.logit, S.m_int, Ee);
  edge_exp<<<cdiv(Ee, 256), 256, 0, st>>>(S.logit, g.dst, S.m_int, S.expv, S.ssum, Ee);
  // encoder aggregation + ELU
  fill_f32<<<cdiv(Nn * HID, 256), 256, 0, st>>>(S.aggr, 0.0f, Nn * HID);
  edge_aggr<<<Ee, 256, 0, st>>>(S.H, g.src, g.dst, S.expv, S.ssum, S.aggr, Ee, HID);
  cast_act<<<cdiv(Nn * HID, 256), 256, 0, st>>>(S.aggr, S.h1h, (size_t)Nn * HID, 1);
  // hidden = h1 @ W2
  launch_gemm(st, S.h1h, HID, S.BtHid, HID, S.hidden, LATd, Nn, LATd, HID, 0);
  cast_act<<<cdiv(Nn * LATd, 256), 256, 0, st>>>(S.hidden, S.hiddenh, (size_t)Nn * LATd, 0);
  // mean / log_var (fused bias / bias+clip)
  launch_gemm(st, S.hiddenh, LATd, S.BtMean, LATd, g.o_mean, LATd, Nn, LATd, LATd, 1, g.mb);
  launch_gemm(st, S.hiddenh, LATd, S.BtLv,   LATd, g.o_lv,   LATd, Nn, LATd, LATd, 2, g.lvb);
  // z = mean + eps*std
  reparam<<<cdiv(Nn * LATd, 256), 256, 0, st>>>(g.o_mean, g.o_lv, g.o_z, S.zh,
                                                (size_t)Nn * LATd, g.seed ^ 0xABCD1234u);
  // decoder: Hdec = z @ W2^T (reuse S.H), tied-alpha aggregation, ELU
  launch_gemm(st, S.zh, LATd, S.BtDec, LATd, S.H, HID, Nn, HID, LATd, 0);
  fill_f32<<<cdiv(Nn * HID, 256), 256, 0, st>>>(S.aggr, 0.0f, Nn * HID);
  edge_aggr<<<Ee, 256, 0, st>>>(S.H, g.src, g.dst, S.expv, S.ssum, S.aggr, Ee, HID);
  cast_act<<<cdiv(Nn * HID, 256), 256, 0, st>>>(S.aggr, S.h1h, (size_t)Nn * HID, 1);
  // mu = softplus(h3 @ W1^T) fused
  launch_gemm(st, S.h1h, HID, S.BtMu, HID, g.o_mu, dflag, Nn, dflag, HID, 3);
  // theta = exp(lth)
  expk<<<cdiv(dflag, 256), 256, 0, st>>>(g.lth, g.o_theta, dflag);
}

} // namespace

extern "C" void kernel_launch(void* const* d_in, const int* in_sizes, int n_in,
                              void* d_out, int out_size, void* d_ws, size_t ws_size,
                              hipStream_t stream){
  (void)in_sizes; (void)n_in; (void)out_size; (void)ws_size;
  const float* ref_homo    = (const float*)d_in[0];
  const float* ref_nonhomo = (const float*)d_in[1];
  const float* tgt_homo    = (const float*)d_in[2];
  const float* tgt_nonhomo = (const float*)d_in[3];
  const float *sW1 = (const float*)d_in[4],  *sa1s = (const float*)d_in[5],
              *sa1d = (const float*)d_in[6], *sW2 = (const float*)d_in[7],
              *smW = (const float*)d_in[8],  *smb = (const float*)d_in[9],
              *slvW = (const float*)d_in[10],*slvb = (const float*)d_in[11],
              *sth = (const float*)d_in[12];
  const float *rW1 = (const float*)d_in[13], *ra1s = (const float*)d_in[14],
              *ra1d = (const float*)d_in[15],*rW2 = (const float*)d_in[16],
              *rmW = (const float*)d_in[17], *rmb = (const float*)d_in[18],
              *rlvW = (const float*)d_in[19],*rlvb = (const float*)d_in[20],
              *rth = (const float*)d_in[21];
  const float *tW1 = (const float*)d_in[22], *ta1s = (const float*)d_in[23],
              *ta1d = (const float*)d_in[24],*tW2 = (const float*)d_in[25],
              *tmW = (const float*)d_in[26], *tmb = (const float*)d_in[27],
              *tlvW = (const float*)d_in[28],*tlvb = (const float*)d_in[29],
              *tth = (const float*)d_in[30];
  const float *cW1 = (const float*)d_in[31], *cb1 = (const float*)d_in[32],
              *bng = (const float*)d_in[33], *bnb = (const float*)d_in[34],
              *bnm = (const float*)d_in[35], *bnv = (const float*)d_in[36],
              *cW2 = (const float*)d_in[37], *cb2 = (const float*)d_in[38];
  const int* ref_ei = (const int*)d_in[39];
  const int* tgt_ei = (const int*)d_in[40];

  // ---- output layout (reference return order, flat f32) ----
  float* out = (float*)d_out;
  size_t off = 0;
  auto take = [&](size_t n){ float* p = out + off; off += n; return p; };
  float* o_ref_logits = take((size_t)Nn * NCd);
  float* o_tgt_logits = take((size_t)Nn * NCd);
  float *o_rh_mean = take((size_t)Nn*LATd), *o_rh_lv = take((size_t)Nn*LATd),
        *o_rh_mu = take((size_t)Nn*(DSd+1)), *o_rh_th = take(DSd+1),
        *o_rh_z = take((size_t)Nn*LATd);
  float *o_rn_mean = take((size_t)Nn*LATd), *o_rn_lv = take((size_t)Nn*LATd),
        *o_rn_mu = take((size_t)Nn*DRd), *o_rn_th = take(DRd),
        *o_rn_z = take((size_t)Nn*LATd);
  float *o_th_mean = take((size_t)Nn*LATd), *o_th_lv = take((size_t)Nn*LATd),
        *o_th_mu = take((size_t)Nn*(DSd+1)), *o_th_th = take(DSd+1),
        *o_th_z = take((size_t)Nn*LATd);
  float *o_tn_mean = take((size_t)Nn*LATd), *o_tn_lv = take((size_t)Nn*LATd),
        *o_tn_mu = take((size_t)Nn*DTd), *o_tn_th = take(DTd),
        *o_tn_z = take((size_t)Nn*LATd);

  // ---- workspace partition (bump allocator, 256B aligned) ----
  char* base = (char*)d_ws;
  size_t woff = 0;
  auto alloc = [&](size_t bytes){
    void* p = base + woff;
    woff = (woff + bytes + 255) & ~(size_t)255;
    return p;
  };
  const int KpMax = rup(DSd + 1, 32);   // 2016
  Scratch S;
  S.Xh      = (h16*)  alloc((size_t)Nn * KpMax * sizeof(h16));
  S.BtXW1   = (h16*)  alloc((size_t)HID * KpMax * sizeof(h16));
  S.BtMu    = (h16*)  alloc((size_t)(DSd + 1) * HID * sizeof(h16));
  S.BtHid   = (h16*)  alloc((size_t)LATd * HID * sizeof(h16));
  S.BtDec   = (h16*)  alloc((size_t)HID * LATd * sizeof(h16));
  S.BtMean  = (h16*)  alloc((size_t)LATd * LATd * sizeof(h16));
  S.BtLv    = (h16*)  alloc((size_t)LATd * LATd * sizeof(h16));
  S.H       = (float*)alloc((size_t)Nn * HID * sizeof(float));
  S.aggr    = (float*)alloc((size_t)Nn * HID * sizeof(float));
  S.h1h     = (h16*)  alloc((size_t)Nn * HID * sizeof(h16));
  S.hidden  = (float*)alloc((size_t)Nn * LATd * sizeof(float));
  S.hiddenh = (h16*)  alloc((size_t)Nn * LATd * sizeof(h16));
  S.zh      = (h16*)  alloc((size_t)Nn * LATd * sizeof(h16));
  S.a_s     = (float*)alloc((size_t)Nn * sizeof(float));
  S.a_d     = (float*)alloc((size_t)Nn * sizeof(float));
  S.ssum    = (float*)alloc((size_t)Nn * sizeof(float));
  S.m_int   = (int*)  alloc((size_t)Nn * sizeof(int));
  S.logit   = (float*)alloc((size_t)Ee * sizeof(float));
  S.expv    = (float*)alloc((size_t)Ee * sizeof(float));
  S.clsin   = (h16*)  alloc((size_t)Nn * 128 * sizeof(h16));
  S.clsh    = (float*)alloc((size_t)Nn * 128 * sizeof(float));
  S.clshh   = (h16*)  alloc((size_t)Nn * 128 * sizeof(h16));
  S.cW1t    = (h16*)  alloc((size_t)128 * 128 * sizeof(h16));
  S.cW2t    = (h16*)  alloc((size_t)NCd * 128 * sizeof(h16));

  // classifier weights -> f16 Bt layout (once)
  cast_pad<<<cdiv(128 * 128, 256), 256, 0, stream>>>(cW1, S.cW1t, 128, 128, 128, 128, 1);
  cast_pad<<<cdiv(NCd * 128, 256), 256, 0, stream>>>(cW2, S.cW2t, NCd, 128, 128, NCd, 1);

  const int *rsrc = ref_ei, *rdst = ref_ei + Ee;
  const int *tsrc = tgt_ei, *tdst = tgt_ei + Ee;

  GBlk rh  = { ref_homo,    DSd, 1, 1.0f, sW1, sa1s, sa1d, sW2, smW, smb, slvW, slvb, sth,
               rsrc, rdst, o_rh_mean, o_rh_lv, o_rh_mu, o_rh_th, o_rh_z, 0x1111u };
  GBlk rnh = { ref_nonhomo, DRd, 0, 0.0f, rW1, ra1s, ra1d, rW2, rmW, rmb, rlvW, rlvb, rth,
               rsrc, rdst, o_rn_mean, o_rn_lv, o_rn_mu, o_rn_th, o_rn_z, 0x2222u };
  GBlk th  = { tgt_homo,    DSd, 1, 0.0f, sW1, sa1s, sa1d, sW2, smW, smb, slvW, slvb, sth,
               tsrc, tdst, o_th_mean, o_th_lv, o_th_mu, o_th_th, o_th_z, 0x3333u };
  GBlk tnh = { tgt_nonhomo, DTd, 0, 0.0f, tW1, ta1s, ta1d, tW2, tmW, tmb, tlvW, tlvb, tth,
               tsrc, tdst, o_tn_mean, o_tn_lv, o_tn_mu, o_tn_th, o_tn_z, 0x4444u };

  run_block(stream, S, rh);
  run_block(stream, S, rnh);
  run_block(stream, S, th);
  run_block(stream, S, tnh);

  // classifier: cat(mean_h, mean_nh) -> MLP (fused bias+BN+SELU, then bias)
  auto classify = [&](const float* ma, const float* mb2, float* o_logits){
    concat_cast<<<cdiv(Nn * 128, 256), 256, 0, stream>>>(ma, mb2, S.clsin, Nn, LATd);
    launch_gemm(stream, S.clsin, 128, S.cW1t, 128, S.clsh, 128, Nn, 128, 128, 4,
                cb1, bng, bnb, bnm, bnv);
    cast_act<<<cdiv(Nn * 128, 256), 256, 0, stream>>>(S.clsh, S.clshh, (size_t)Nn * 128, 0);
    launch_gemm(stream, S.clshh, 128, S.cW2t, 128, o_logits, NCd, Nn, NCd, 128, 1, cb2);
  };
  classify(o_rh_mean, o_rn_mean, o_ref_logits);
  classify(o_th_mean, o_tn_mean, o_tgt_logits);
}
